// TransFusionHead_81716047774391
// MI455X (gfx1250) — matrix operand order repaired and checked by reading, not compiled, not run on hardware
//
#include <hip/hip_runtime.h>

// ---------------------------------------------------------------------------
// TransFusion head for MI455X (gfx1250, wave32).
// All heavy contractions -> v_wmma_f32_16x16x32_f16 (f16 in, f32 accum).
//  - 3x3 convs as implicit GEMM; A (weight) tiles DMA'd to LDS by the Tensor
//    Data Mover (double-buffered, s_wait_tensorcnt), B tiles staged transposed
//    in LDS so fragments are contiguous ds_load_b128.
//  - dense layers via generic token-major WMMA GEMM
//  - attention as flash-style streaming (head_dim = 16 = WMMA tile)
// ---------------------------------------------------------------------------

typedef __attribute__((ext_vector_type(16))) _Float16 v16h;
typedef __attribute__((ext_vector_type(8)))  _Float16 v8h;
typedef __attribute__((ext_vector_type(8)))  float    v8f;
typedef __attribute__((ext_vector_type(4)))  float    v4f;

#define PW 200          // padded image width  (180 + pads + tile slack)
#define PH 182          // padded image height (180 + 2)
#define PSLAB (PH*PW)   // 36400
#define HWSZ 32400      // 180*180
#define NPQ 200         // proposals

#if defined(__gfx1250__) && __has_builtin(__builtin_amdgcn_tensor_load_to_lds) && \
    __has_builtin(__builtin_amdgcn_s_wait_tensorcnt)
#define HAVE_TDM 1
#else
#define HAVE_TDM 0
#endif

#if HAVE_TDM
typedef __attribute__((ext_vector_type(4))) unsigned int v4u;
typedef __attribute__((ext_vector_type(8))) int          v8i;
typedef __attribute__((ext_vector_type(4))) int          v4i;

// Issue a 2-D TDM tile load: tile_x f16 elements per row, tile_y rows with a
// row stride of stride_elems, from gaddr into LDS byte offset lds_off.
// D# layout per CDNA5 ISA ch.8 (group0: count/lds/global/type, group1: dims).
__device__ __forceinline__ void tdm_load_2d(unsigned lds_off, const void* gaddr,
                                            unsigned tile_x, unsigned tile_y,
                                            unsigned stride_elems) {
  unsigned long long ga = (unsigned long long)gaddr;
  v4u g0;
  g0.x = 1u;                                              // count=1, user D#
  g0.y = lds_off;                                         // lds_addr [63:32]
  g0.z = (unsigned)ga;                                    // global_addr [95:64]
  g0.w = ((unsigned)(ga >> 32) & 0x01FFFFFFu) | (2u << 30); // [120:96] | type=2
  const unsigned td0 = 1u << 20, td1 = 1u << 20;          // generous tensor dims
  v8i g1;
  g1[0] = (int)(1u << 16);                                // data_size=1 (2B)
  g1[1] = (int)((td0 & 0xFFFFu) << 16);                   // tensor_dim0 lo
  g1[2] = (int)((td0 >> 16) | ((td1 & 0xFFFFu) << 16));   // dim0 hi | dim1 lo
  g1[3] = (int)((td1 >> 16) | (tile_x << 16));            // dim1 hi | tile_dim0
  g1[4] = (int)(tile_y & 0xFFFFu);                        // tile_dim1 (dim2=0)
  g1[5] = (int)stride_elems;                              // tensor_dim0_stride
  g1[6] = 0;
  g1[7] = 0;
  v4i z4 = {0, 0, 0, 0};
#if __clang_major__ >= 23
  v8i z8 = {0, 0, 0, 0, 0, 0, 0, 0};
  __builtin_amdgcn_tensor_load_to_lds(g0, g1, z4, z4, z8, 0);
#else
  __builtin_amdgcn_tensor_load_to_lds(g0, g1, z4, z4, 0);
#endif
}
#endif  // HAVE_TDM

// ======================= prep / convert kernels ============================

__global__ __launch_bounds__(256) void k_pad_in(const float* __restrict__ src,
                                                _Float16* __restrict__ dst, int C) {
  size_t i = (size_t)blockIdx.x * 256 + threadIdx.x;
  size_t tot = (size_t)C * PSLAB;
  if (i >= tot) return;
  int c = (int)(i / PSLAB);
  int p = (int)(i % PSLAB);
  int y = p / PW, x = p % PW;
  float v = 0.f;
  if (y >= 1 && y <= 180 && x >= 1 && x <= 180)
    v = src[((size_t)c * 180 + (y - 1)) * 180 + (x - 1)];
  dst[i] = (_Float16)v;
}

__global__ __launch_bounds__(256) void k_zero_h(_Float16* p, size_t n) {
  size_t i = (size_t)blockIdx.x * 256 + threadIdx.x;
  if (i < n) p[i] = (_Float16)0.f;
}

// dense weight convert f32 -> f16 row-major [Md][K], zero-padding rows >= Ms
__global__ __launch_bounds__(256) void k_cvt_w(const float* __restrict__ s,
                                               _Float16* __restrict__ d,
                                               int Ms, int Md, int K) {
  int i = blockIdx.x * 256 + threadIdx.x;
  if (i >= Md * K) return;
  int m = i / K, k = i - m * K;
  d[i] = (m < Ms) ? (_Float16)s[(size_t)m * K + k] : (_Float16)0.f;
}

// conv weight relayout: [Co][Ci][3][3] f32 -> [Cod][9][Cin] f16 (pad rows)
__global__ __launch_bounds__(256) void k_cvt_convw(const float* __restrict__ s,
                                                   _Float16* __restrict__ d,
                                                   int Cos, int Cod, int Cin) {
  int i = blockIdx.x * 256 + threadIdx.x;
  int tot = Cod * 9 * Cin;
  if (i >= tot) return;
  int co = i / (9 * Cin);
  int rem = i - co * 9 * Cin;
  int o = rem / Cin, ci = rem - o * Cin;
  d[i] = (co < Cos) ? (_Float16)s[((size_t)co * Cin + ci) * 9 + o] : (_Float16)0.f;
}

__global__ void k_padf(const float* __restrict__ s, float* __restrict__ d, int Ms, int Md) {
  int i = blockIdx.x * blockDim.x + threadIdx.x;
  if (i < Md) d[i] = (i < Ms) ? s[i] : 0.f;
}

__global__ void k_bnscale(const float* __restrict__ g, float* __restrict__ d, int n) {
  int i = blockIdx.x * blockDim.x + threadIdx.x;
  if (i < n) d[i] = g[i] * rsqrtf(1.f + 1e-5f);
}

// ===================== 3x3 conv as implicit WMMA GEMM ======================
// Workgroup tile: M = Cout (<=128), N = 32 positions. 8 waves each own one
// 16-row M slice and run 2 WMMAs/step over the two 16-col N subtiles.
// A tile (Cout x 32 f16, row stride 9*Cin) is DMA'd to LDS via the TDM,
// double buffered; B tile staged transposed ([n][k]) for b128 fragment loads.
__global__ __launch_bounds__(256) void k_conv3x3(
    const _Float16* __restrict__ inp,   // [Cin][PH][PW] padded
    const _Float16* __restrict__ wts,   // [Cout][9][Cin]
    const float* __restrict__ scale,    // [Cout] or null (BN scale)
    const float* __restrict__ shift,    // [Cout] or null (bias / BN shift)
    float* __restrict__ out32,          // null, or see chmaj
    _Float16* __restrict__ out16,       // null, or [Cout][PH][PW] padded
    int Cin, int Cout, int mstore, int flags, int ld32, int chmaj) {
  __shared__ _Float16 at[2 * 128 * 32];   // A double buffer (TDM destination)
  __shared__ _Float16 bt[32 * 40];        // B tile, [n][k], padded stride 40
  const int tid = threadIdx.x, lane = tid & 31, wid = tid >> 5;
  const int hf = lane >> 4, r = lane & 15;
  const int y0 = blockIdx.x / 6, x0 = (blockIdx.x % 6) * 32;
  const int mtiles = Cout >> 4;
  const int mt = wid % mtiles;
  const int skk = tid >> 3, spart = tid & 7;   // B staging: k row, 4-col group
  const int csteps = Cin >> 5;
  const int nsteps = 9 * csteps;
  const int arow_stride = 9 * Cin;
  v8f acc0 = {}, acc1 = {};
#if HAVE_TDM
  const unsigned at_off = (unsigned)(unsigned long long)(const void*)&at[0];
  if (wid == 0)
    tdm_load_2d(at_off, wts, 32, (unsigned)Cout, (unsigned)arow_stride);
#endif
  for (int s = 0; s < nsteps; ++s) {
    const int o = s / csteps, cc = (s - o * csteps) << 5;
    const int dy = o / 3, dx = o - dy * 3;
    const _Float16* brow = inp + (size_t)(y0 + dy) * PW + (x0 + dx) + (size_t)cc * PSLAB;
    __syncthreads();                       // previous step's LDS reads done
    {                                      // stage B: 32 k rows x 32 n cols
      const _Float16* g = brow + (size_t)skk * PSLAB + spart * 4;
#pragma unroll
      for (int j = 0; j < 4; ++j)
        bt[(spart * 4 + j) * 40 + skk] = g[j];
      if (s + 1 < nsteps)
        __builtin_prefetch(g + (size_t)32 * PSLAB, 0, 3);
    }
#if HAVE_TDM
    if (wid == 0) __builtin_amdgcn_s_wait_tensorcnt(0);  // A[s&1] complete
#else
    for (int c = tid; c < Cout * 4; c += 256)            // cooperative A copy
      ((v8h*)at)[c] = *(const v8h*)(wts + (size_t)(c >> 2) * arow_stride +
                                    (size_t)o * Cin + cc + (c & 3) * 8);
#endif
    __syncthreads();                       // A + B visible to all waves
#if HAVE_TDM
    if (wid == 0 && s + 1 < nsteps) {      // prefetch next A tile via TDM
      const int o2 = (s + 1) / csteps, cc2 = ((s + 1) - o2 * csteps) << 5;
      tdm_load_2d(at_off + (unsigned)(((s + 1) & 1) * Cout * 32 * 2),
                  wts + (size_t)o2 * Cin + cc2, 32, (unsigned)Cout,
                  (unsigned)arow_stride);
    }
    const _Float16* abuf = at + (size_t)(s & 1) * Cout * 32;
#else
    const _Float16* abuf = at;
#endif
    v16h af, bf0, bf1;
    const _Float16* ap = abuf + (size_t)(mt * 16 + r) * 32 + hf * 16;
    ((v8h*)&af)[0] = *(const v8h*)ap;
    ((v8h*)&af)[1] = *(const v8h*)(ap + 8);
    const _Float16* bp0 = &bt[r * 40 + hf * 16];
    ((v8h*)&bf0)[0] = *(const v8h*)bp0;
    ((v8h*)&bf0)[1] = *(const v8h*)(bp0 + 8);
    const _Float16* bp1 = &bt[(16 + r) * 40 + hf * 16];
    ((v8h*)&bf1)[0] = *(const v8h*)bp1;
    ((v8h*)&bf1)[1] = *(const v8h*)(bp1 + 8);
    acc0 = __builtin_amdgcn_wmma_f32_16x16x32_f16(false, af, false, bf0,
                                                  (short)0, acc0, false, false);
    acc1 = __builtin_amdgcn_wmma_f32_16x16x32_f16(false, af, false, bf1,
                                                  (short)0, acc1, false, false);
  }
#pragma unroll
  for (int t2 = 0; t2 < 2; ++t2) {
    const v8f acc = t2 ? acc1 : acc0;
    const int x = x0 + t2 * 16 + r;
#pragma unroll
    for (int j = 0; j < 8; ++j) {
      const int m = mt * 16 + j + hf * 8;
      float v = acc[j];
      if (scale) v *= scale[m];
      if (shift) v += shift[m];
      if (flags & 1) v = fmaxf(v, 0.f);
      if (flags & 2) v = 1.f / (1.f + __expf(-v));
      if (x < 180 && m < mstore) {
        if (out32) {
          if (chmaj) out32[(size_t)m * HWSZ + y0 * 180 + x] = v;
          else       out32[(size_t)(y0 * 180 + x) * ld32 + m] = v;
        }
        if (out16) out16[(size_t)m * PSLAB + (y0 + 1) * PW + (x + 1)] = (_Float16)v;
      }
    }
  }
}

// ================== generic token-major WMMA GEMM ==========================
// out[n][m] = f( sum_k W16[m][k] * act[n][k] ),  M mult of 16, K mult of 32.
__global__ __launch_bounds__(256) void k_gemm(
    const _Float16* __restrict__ W, const float* __restrict__ act,
    float* __restrict__ out, const float* __restrict__ scale,
    const float* __restrict__ shift, int M, int N, int K,
    int lda, int ldc, int relu) {
  const int lane = threadIdx.x & 31, wid = threadIdx.x >> 5;
  const int hf = lane >> 4, r = lane & 15;
  const int mtiles = M >> 4;
  const int ntiles = (N + 15) >> 4;
  const int tile = blockIdx.x * 8 + wid;
  if (tile >= mtiles * ntiles) return;
  const int mt = tile % mtiles, nt = tile / mtiles;
  const int n = nt * 16 + r;
  const int ncl = n < N ? n : N - 1;
  const _Float16* wrow = W + (size_t)(mt * 16 + r) * K;
  const float* arow = act + (size_t)ncl * lda;
  v8f acc = {};
  for (int k0 = 0; k0 < K; k0 += 32) {
    v16h af, bf;
    const _Float16* wp = wrow + k0 + hf * 16;
    ((v8h*)&af)[0] = *(const v8h*)wp;
    ((v8h*)&af)[1] = *(const v8h*)(wp + 8);
    const v4f* ap = (const v4f*)(arow + k0 + hf * 16);
#pragma unroll
    for (int q = 0; q < 4; ++q) {
      v4f fv = ap[q];
#pragma unroll
      for (int e = 0; e < 4; ++e) bf[q * 4 + e] = (_Float16)fv[e];
    }
    acc = __builtin_amdgcn_wmma_f32_16x16x32_f16(false, af, false, bf,
                                                 (short)0, acc, false, false);
  }
#pragma unroll
  for (int j = 0; j < 8; ++j) {
    const int mm = mt * 16 + j + hf * 8;
    const int nn = nt * 16 + r;
    if (nn < N) {
      float v = acc[j];
      if (scale) v *= scale[mm];
      if (shift) v += shift[mm];
      if (relu) v = fmaxf(v, 0.f);
      out[(size_t)nn * ldc + mm] = v;
    }
  }
}

// ===================== flash attention (head_dim = 16) =====================
// 8 waves = 8 heads; 1 wave handles a 16-query block, streams keys in 32s.
// S^T = K x Q^T so the D-layout of S^T is exactly the A-fragment of P x V.
__global__ __launch_bounds__(256) void k_attn(
    const float* __restrict__ Qp, const float* __restrict__ Kp,
    const float* __restrict__ Vp, float* __restrict__ outp, int Nq, int Nk) {
  const int lane = threadIdx.x & 31, head = threadIdx.x >> 5;
  const int hf = lane >> 4, r = lane & 15;
  const int qblk = blockIdx.x;
  const int qrow = qblk * 16 + r;
  const int qc = qrow < Nq ? qrow : Nq - 1;
  // Q as B-fragment for S^T: lanes<16 hold d 0..15, lanes>=16 hold pad zeros.
  v16h qf = {};
  if (hf == 0) {
    const v4f* qp = (const v4f*)(Qp + (size_t)qc * 128 + head * 16);
#pragma unroll
    for (int q4 = 0; q4 < 4; ++q4) {
      v4f fv = qp[q4];
#pragma unroll
      for (int e = 0; e < 4; ++e) qf[q4 * 4 + e] = (_Float16)(fv[e] * 0.25f);
    }
  }
  v8f o0 = {};
  float mrun = -1e30f, lrun = 0.f;
  const int nblk = (Nk + 31) >> 5;
  for (int kb = 0; kb < nblk; ++kb) {
    const int kbase = kb << 5;
    v8f s0, s1;
#pragma unroll
    for (int g = 0; g < 2; ++g) {
      const int key = kbase + g * 16 + r;
      const int kc = key < Nk ? key : Nk - 1;
      v16h kf = {};   // K as A-fragment (d padded to 32 with zeros)
      const float* kp = Kp + (size_t)kc * 128 + head * 16;
#pragma unroll
      for (int e = 0; e < 16; ++e) kf[e] = hf ? (_Float16)0.f : (_Float16)kp[e];
      v8f z = {};
      v8f sg = __builtin_amdgcn_wmma_f32_16x16x32_f16(false, kf, false, qf,
                                                      (short)0, z, false, false);
#pragma unroll
      for (int j = 0; j < 8; ++j) {
        const int key_j = kbase + g * 16 + j + hf * 8;
        if (key_j >= Nk) sg[j] = -1e30f;
      }
      if (g == 0) s0 = sg; else s1 = sg;
    }
    // per-query (q = r) stats over the 32-key block
    float lm = -1e30f;
#pragma unroll
    for (int j = 0; j < 8; ++j) lm = fmaxf(lm, fmaxf(s0[j], s1[j]));
    lm = fmaxf(lm, __shfl_xor(lm, 16));
    const float mnew = fmaxf(mrun, lm);
    const float corr = __expf(mrun - mnew);
    v16h pf;
    float ls = 0.f;
#pragma unroll
    for (int j = 0; j < 8; ++j) {
      float p0 = __expf(s0[j] - mnew);
      float p1 = __expf(s1[j] - mnew);
      ls += p0 + p1;
      pf[j] = (_Float16)p0;
      pf[8 + j] = (_Float16)p1;
    }
    ls += __shfl_xor(ls, 16);
    lrun = lrun * corr + ls;
    mrun = mnew;
#pragma unroll
    for (int j = 0; j < 8; ++j) o0[j] *= __shfl(corr, j + hf * 8);
    // V as B-fragment: col d = r, keys hf*8+e (e<8) and 16+hf*8+(e-8) (e>=8)
    v16h vf;
#pragma unroll
    for (int e = 0; e < 16; ++e) {
      int key = kbase + hf * 8 + (e & 7) + (e >> 3) * 16;
      int kc = key < Nk ? key : Nk - 1;
      vf[e] = (_Float16)Vp[(size_t)kc * 128 + head * 16 + r];
    }
    o0 = __builtin_amdgcn_wmma_f32_16x16x32_f16(false, pf, false, vf,
                                                (short)0, o0, false, false);
  }
#pragma unroll
  for (int j = 0; j < 8; ++j) {
    const int q = j + hf * 8;
    const float inv = 1.f / __shfl(lrun, q);
    const int qg = qblk * 16 + q;
    if (qg < Nq) outp[(size_t)qg * 128 + head * 16 + r] = o0[j] * inv;
  }
}

// =========================== misc pipeline =================================

__global__ __launch_bounds__(256) void k_nms(const float* __restrict__ hm,
                                             float* __restrict__ nmsout,
                                             float* __restrict__ scratch) {
  int i = blockIdx.x * 256 + threadIdx.x;
  if (i >= 10 * HWSZ) return;
  int c = i / HWSZ, p = i - c * HWSZ;
  int y = p / 180, x = p - y * 180;
  float v = hm[i];
  float lm;
  if (c >= 8) lm = v;
  else if (y == 0 || y == 179 || x == 0 || x == 179) lm = 0.f;
  else {
    lm = -1e30f;
#pragma unroll
    for (int dy = -1; dy <= 1; ++dy)
#pragma unroll
      for (int dx = -1; dx <= 1; ++dx)
        lm = fmaxf(lm, hm[(size_t)c * HWSZ + (y + dy) * 180 + (x + dx)]);
  }
  float rv = (v == lm) ? v : 0.f;
  nmsout[i] = rv;
  scratch[i] = rv;
}

__global__ __launch_bounds__(1024) void k_topk(float* __restrict__ scr,
                                               int* __restrict__ cls,
                                               int* __restrict__ pos) {
  __shared__ float sv[1024];
  __shared__ int si[1024];
  const int tid = threadIdx.x;
  for (int t = 0; t < NPQ; ++t) {
    float best = -3e38f; int bi = 0x7fffffff;
    for (int i = tid; i < 10 * HWSZ; i += 1024) {
      float v = scr[i];
      if (v > best || (v == best && i < bi)) { best = v; bi = i; }
    }
    sv[tid] = best; si[tid] = bi;
    __syncthreads();
    for (int s = 512; s > 0; s >>= 1) {
      if (tid < s) {
        float ov = sv[tid + s]; int oi = si[tid + s];
        if (ov > sv[tid] || (ov == sv[tid] && oi < si[tid])) { sv[tid] = ov; si[tid] = oi; }
      }
      __syncthreads();
    }
    if (tid == 0) {
      int idx = si[0];
      cls[t] = idx / HWSZ;
      pos[t] = idx - (idx / HWSZ) * HWSZ;
      scr[idx] = -3e38f;
    }
    __syncthreads();
  }
}

__global__ __launch_bounds__(256) void k_build_q(const float* __restrict__ lidar_tm,
                                                 const float* __restrict__ cls_w,
                                                 const float* __restrict__ cls_b,
                                                 const int* __restrict__ cls,
                                                 const int* __restrict__ pos,
                                                 float* __restrict__ qt) {
  int i = blockIdx.x * 256 + threadIdx.x;
  if (i >= NPQ * 128) return;
  int q = i >> 7, c = i & 127;
  qt[i] = lidar_tm[(size_t)pos[q] * 128 + c] + cls_w[c * 10 + cls[q]] + cls_b[c];
}

__global__ void k_build_qpos(const int* __restrict__ cls, const int* __restrict__ pos,
                             const float* __restrict__ hmnms,
                             float* __restrict__ qpos, float* __restrict__ qsc) {
  int i = blockIdx.x * blockDim.x + threadIdx.x;
  if (i >= NPQ) return;
  int p = pos[i];
  qpos[i * 2 + 0] = (float)(p % 180) + 0.5f;   // bev flipped: (y-part, x-part)
  qpos[i * 2 + 1] = (float)(p / 180) + 0.5f;
  qsc[i] = hmnms[(size_t)cls[i] * HWSZ + p];
}

// pos-embed layer1: relu(bn(xy @ w1.T + b1)) ; isbev synthesizes bev coords
__global__ __launch_bounds__(256) void k_pe1(const float* __restrict__ xy,
                                             const float* __restrict__ w1,
                                             const float* __restrict__ b1,
                                             const float* __restrict__ scale,
                                             const float* __restrict__ bb,
                                             float* __restrict__ out, int N, int isbev) {
  int i = blockIdx.x * 256 + threadIdx.x;
  if (i >= N * 128) return;
  int n = i >> 7, m = i & 127;
  float x, y;
  if (isbev) { x = (float)(n % 180) + 0.5f; y = (float)(n / 180) + 0.5f; }
  else { x = xy[n * 2]; y = xy[n * 2 + 1]; }
  float t = w1[m * 2] * x + w1[m * 2 + 1] * y + b1[m];
  out[i] = fmaxf(t * scale[m] + bb[m], 0.f);
}

__global__ __launch_bounds__(256) void k_vadd(const float* __restrict__ a,
                                              const float* __restrict__ b,
                                              float* __restrict__ c, size_t n) {
  size_t i = (size_t)blockIdx.x * 256 + threadIdx.x;
  if (i < n) c[i] = a[i] + b[i];
}

__global__ __launch_bounds__(256) void k_add_ln(const float* __restrict__ a,
                                                const float* __restrict__ b,
                                                const float* __restrict__ g,
                                                const float* __restrict__ be,
                                                float* __restrict__ out, int N) {
  const int lane = threadIdx.x & 31, wid = threadIdx.x >> 5;
  const int tok = blockIdx.x * 8 + wid;
  if (tok >= N) return;
  const float* pa = a + (size_t)tok * 128;
  const float* pb = b + (size_t)tok * 128;
  float v[4];
  float s = 0.f;
#pragma unroll
  for (int e = 0; e < 4; ++e) { v[e] = pa[lane * 4 + e] + pb[lane * 4 + e]; s += v[e]; }
#pragma unroll
  for (int d = 1; d < 32; d <<= 1) s += __shfl_xor(s, d);
  const float mean = s * (1.f / 128.f);
  float q = 0.f;
#pragma unroll
  for (int e = 0; e < 4; ++e) { float t = v[e] - mean; q += t * t; }
#pragma unroll
  for (int d = 1; d < 32; d <<= 1) q += __shfl_xor(q, d);
  const float inv = rsqrtf(q * (1.f / 128.f) + 1e-5f);
#pragma unroll
  for (int e = 0; e < 4; ++e)
    out[(size_t)tok * 128 + lane * 4 + e] =
        (v[e] - mean) * inv * g[lane * 4 + e] + be[lane * 4 + e];
}

__global__ void k_decode(const float* __restrict__ ctr, const float* __restrict__ hei,
                         const float* __restrict__ dm, const float* __restrict__ rot,
                         const float* __restrict__ vel, const float* __restrict__ hmh,
                         const float* __restrict__ qpos, const float* __restrict__ qsc,
                         const int* __restrict__ cls, float* __restrict__ out) {
  int i = blockIdx.x * blockDim.x + threadIdx.x;
  if (i >= NPQ) return;
  float cx = (ctr[i * 16 + 0] + qpos[i * 2 + 0]) * 0.6f - 54.0f;
  float cy = (ctr[i * 16 + 1] + qpos[i * 2 + 1]) * 0.6f - 54.0f;
  float hh = hei[i * 16 + 0];
  float d0 = __expf(dm[i * 16 + 0]);
  float d1 = __expf(dm[i * 16 + 1]);
  float d2 = __expf(dm[i * 16 + 2]);
  float yaw = atan2f(rot[i * 16 + 0], rot[i * 16 + 1]);
  float sc = (1.f / (1.f + __expf(-hmh[i * 16 + cls[i]]))) * qsc[i];
  int lab = sc > 0.f ? cls[i] : 0;
  out[i * 9 + 0] = cx;  out[i * 9 + 1] = cy;  out[i * 9 + 2] = hh;
  out[i * 9 + 3] = d0;  out[i * 9 + 4] = d1;  out[i * 9 + 5] = d2;
  out[i * 9 + 6] = yaw;
  out[i * 9 + 7] = vel[i * 16 + 0];
  out[i * 9 + 8] = vel[i * 16 + 1];
  out[9 * NPQ + i] = sc;
  out[10 * NPQ + i] = (float)lab;
}

// =============================== host ======================================

static inline int cdiv(long long a, long long b) { return (int)((a + b - 1) / b); }

extern "C" void kernel_launch(void* const* d_in, const int* in_sizes, int n_in,
                              void* d_out, int out_size, void* d_ws, size_t ws_size,
                              hipStream_t stream) {
  (void)in_sizes; (void)n_in; (void)out_size; (void)ws_size;
  // ---- inputs (JAX pytree sorted-key order; d_in[0] = feats) ----
#define PF(i) ((const float*)d_in[(i)])
  const float* feats = PF(0);
  const float *ca_bk = PF(1), *ca_bo = PF(2), *ca_bq = PF(3), *ca_bv = PF(4);
  const float *ca_wk = PF(5), *ca_wo = PF(6), *ca_wq = PF(7), *ca_wv = PF(8);
  const float *cls_b = PF(9), *cls_w = PF(10);
  const float *ffn_b1 = PF(11), *ffn_b2 = PF(12), *ffn_w1 = PF(13), *ffn_w2 = PF(14);
  // heads, sorted: center, dim, heatmap, height, rot, vel; each b,b2,g,w1,w2
  const int head_odim[6] = {2, 3, 10, 1, 2, 2};
  const float *h_b[6], *h_b2[6], *h_g[6], *h_w1[6], *h_w2[6];
  for (int j = 0; j < 6; ++j) {
    int base = 15 + j * 5;
    h_b[j] = PF(base); h_b2[j] = PF(base + 1); h_g[j] = PF(base + 2);
    h_w1[j] = PF(base + 3); h_w2[j] = PF(base + 4);
  }
  const float *hm1_b = PF(45), *hm1_g = PF(46), *hm1_w = PF(47);
  const float *hm2_b = PF(48), *hm2_w = PF(49);
  const float *ln1_b = PF(50), *ln1_g = PF(51), *ln2_b = PF(52), *ln2_g = PF(53);
  const float *ln3_b = PF(54), *ln3_g = PF(55);
  const float *pc_b = PF(56), *pc_b1 = PF(57), *pc_b2 = PF(58), *pc_g = PF(59);
  const float *pc_w1 = PF(60), *pc_w2 = PF(61);
  const float *ps_b = PF(62), *ps_b1 = PF(63), *ps_b2 = PF(64), *ps_g = PF(65);
  const float *ps_w1 = PF(66), *ps_w2 = PF(67);
  const float *sa_bk = PF(68), *sa_bo = PF(69), *sa_bq = PF(70), *sa_bv = PF(71);
  const float *sa_wk = PF(72), *sa_wo = PF(73), *sa_wq = PF(74), *sa_wv = PF(75);
  const float *shared_b = PF(76), *shared_w = PF(77);

  // ---- workspace carve-out (needs ~165 MB) ----
  char* wsb = (char*)d_ws;
  size_t off = 0;
  auto A = [&](size_t bytes) { void* p = wsb + off; off = (off + bytes + 255) & ~(size_t)255; return p; };
  _Float16* pad512 = (_Float16*)A((size_t)512 * PSLAB * 2);
  _Float16* padA   = (_Float16*)A((size_t)128 * PSLAB * 2);
  _Float16* padB   = (_Float16*)A((size_t)128 * PSLAB * 2);
  _Float16* w16_sh = (_Float16*)A((size_t)128 * 9 * 512 * 2);
  _Float16* w16_h1 = (_Float16*)A((size_t)128 * 9 * 128 * 2);
  _Float16* w16_h2 = (_Float16*)A((size_t)16 * 9 * 128 * 2);
  float* lidar_tm = (float*)A((size_t)HWSZ * 128 * 4);
  float* hm       = (float*)A((size_t)10 * HWSZ * 4);
  float* hmnms    = (float*)A((size_t)10 * HWSZ * 4);
  float* tscr     = (float*)A((size_t)10 * HWSZ * 4);
  int* cidx = (int*)A(NPQ * 4);
  int* pidx = (int*)A(NPQ * 4);
  float* kpe   = (float*)A((size_t)HWSZ * 128 * 4);
  float* kvin  = (float*)A((size_t)HWSZ * 128 * 4);
  float* kproj = (float*)A((size_t)HWSZ * 128 * 4);
  float* vproj = (float*)A((size_t)HWSZ * 128 * 4);
  float* bevh  = (float*)A((size_t)HWSZ * 128 * 4);
  float* qt   = (float*)A(NPQ * 128 * 4);
  float* qpos = (float*)A(NPQ * 2 * 4);
  float* qsc  = (float*)A(NPQ * 4);
  float* qpeh = (float*)A(NPQ * 128 * 4);
  float* qpe  = (float*)A(NPQ * 128 * 4);
  float* q0   = (float*)A(NPQ * 128 * 4);
  float* sq   = (float*)A(NPQ * 128 * 4);
  float* sk   = (float*)A(NPQ * 128 * 4);
  float* sv   = (float*)A(NPQ * 128 * 4);
  float* ao   = (float*)A(NPQ * 128 * 4);
  float* po   = (float*)A(NPQ * 128 * 4);
  float* q1   = (float*)A(NPQ * 128 * 4);
  float* qin  = (float*)A(NPQ * 128 * 4);
  float* cq   = (float*)A(NPQ * 128 * 4);
  float* q2   = (float*)A(NPQ * 128 * 4);
  float* f1   = (float*)A(NPQ * 256 * 4);
  float* f2   = (float*)A(NPQ * 128 * 4);
  float* q3   = (float*)A(NPQ * 128 * 4);
  float* hid1 = (float*)A(NPQ * 64 * 4);
  float* hout[6];
  for (int j = 0; j < 6; ++j) hout[j] = (float*)A(NPQ * 16 * 4);
  _Float16* g_psw2 = (_Float16*)A(128 * 128 * 2);
  _Float16* g_pcw2 = (_Float16*)A(128 * 128 * 2);
  _Float16* g_sawq = (_Float16*)A(128 * 128 * 2);
  _Float16* g_sawk = (_Float16*)A(128 * 128 * 2);
  _Float16* g_sawv = (_Float16*)A(128 * 128 * 2);
  _Float16* g_sawo = (_Float16*)A(128 * 128 * 2);
  _Float16* g_cawq = (_Float16*)A(128 * 128 * 2);
  _Float16* g_cawk = (_Float16*)A(128 * 128 * 2);
  _Float16* g_cawv = (_Float16*)A(128 * 128 * 2);
  _Float16* g_cawo = (_Float16*)A(128 * 128 * 2);
  _Float16* g_ffn1 = (_Float16*)A(256 * 128 * 2);
  _Float16* g_ffn2 = (_Float16*)A(128 * 256 * 2);
  _Float16* g_hw1[6]; _Float16* g_hw2[6];
  for (int j = 0; j < 6; ++j) { g_hw1[j] = (_Float16*)A(64 * 128 * 2); g_hw2[j] = (_Float16*)A(16 * 64 * 2); }
  float* hm1_sc = (float*)A(128 * 4);
  float* ps_sc  = (float*)A(128 * 4);
  float* pc_sc  = (float*)A(128 * 4);
  float* hg_sc[6]; float* hb2p[6];
  for (int j = 0; j < 6; ++j) { hg_sc[j] = (float*)A(64 * 4); hb2p[j] = (float*)A(16 * 4); }
  float* hm2bp = (float*)A(16 * 4);

  float* out = (float*)d_out;

  // ---- prep: pad/convert activations & weights ----
  k_pad_in<<<cdiv((long long)512 * PSLAB, 256), 256, 0, stream>>>(feats, pad512, 512);
  k_zero_h<<<cdiv((long long)128 * PSLAB, 256), 256, 0, stream>>>(padA, (size_t)128 * PSLAB);
  k_zero_h<<<cdiv((long long)128 * PSLAB, 256), 256, 0, stream>>>(padB, (size_t)128 * PSLAB);
  k_cvt_convw<<<cdiv(128 * 9 * 512, 256), 256, 0, stream>>>(shared_w, w16_sh, 128, 128, 512);
  k_cvt_convw<<<cdiv(128 * 9 * 128, 256), 256, 0, stream>>>(hm1_w, w16_h1, 128, 128, 128);
  k_cvt_convw<<<cdiv(16 * 9 * 128, 256), 256, 0, stream>>>(hm2_w, w16_h2, 10, 16, 128);
  k_bnscale<<<1, 128, 0, stream>>>(hm1_g, hm1_sc, 128);
  k_bnscale<<<1, 128, 0, stream>>>(ps_g, ps_sc, 128);
  k_bnscale<<<1, 128, 0, stream>>>(pc_g, pc_sc, 128);
  k_padf<<<1, 16, 0, stream>>>(hm2_b, hm2bp, 10, 16);
  for (int j = 0; j < 6; ++j) {
    k_bnscale<<<1, 64, 0, stream>>>(h_g[j], hg_sc[j], 64);
    k_padf<<<1, 16, 0, stream>>>(h_b2[j], hb2p[j], head_odim[j], 16);
    k_cvt_w<<<cdiv(64 * 128, 256), 256, 0, stream>>>(h_w1[j], g_hw1[j], 64, 64, 128);
    k_cvt_w<<<cdiv(16 * 64, 256), 256, 0, stream>>>(h_w2[j], g_hw2[j], head_odim[j], 16, 64);
  }
  k_cvt_w<<<cdiv(128 * 128, 256), 256, 0, stream>>>(ps_w2, g_psw2, 128, 128, 128);
  k_cvt_w<<<cdiv(128 * 128, 256), 256, 0, stream>>>(pc_w2, g_pcw2, 128, 128, 128);
  k_cvt_w<<<cdiv(128 * 128, 256), 256, 0, stream>>>(sa_wq, g_sawq, 128, 128, 128);
  k_cvt_w<<<cdiv(128 * 128, 256), 256, 0, stream>>>(sa_wk, g_sawk, 128, 128, 128);
  k_cvt_w<<<cdiv(128 * 128, 256), 256, 0, stream>>>(sa_wv, g_sawv, 128, 128, 128);
  k_cvt_w<<<cdiv(128 * 128, 256), 256, 0, stream>>>(sa_wo, g_sawo, 128, 128, 128);
  k_cvt_w<<<cdiv(128 * 128, 256), 256, 0, stream>>>(ca_wq, g_cawq, 128, 128, 128);
  k_cvt_w<<<cdiv(128 * 128, 256), 256, 0, stream>>>(ca_wk, g_cawk, 128, 128, 128);
  k_cvt_w<<<cdiv(128 * 128, 256), 256, 0, stream>>>(ca_wv, g_cawv, 128, 128, 128);
  k_cvt_w<<<cdiv(128 * 128, 256), 256, 0, stream>>>(ca_wo, g_cawo, 128, 128, 128);
  k_cvt_w<<<cdiv(256 * 128, 256), 256, 0, stream>>>(ffn_w1, g_ffn1, 256, 256, 128);
  k_cvt_w<<<cdiv(128 * 256, 256), 256, 0, stream>>>(ffn_w2, g_ffn2, 128, 128, 256);

  // ---- conv stack (implicit WMMA GEMM, TDM-fed A tiles) ----
  k_conv3x3<<<1080, 256, 0, stream>>>(pad512, w16_sh, nullptr, shared_b,
                                      lidar_tm, padA, 512, 128, 128, 0, 128, 0);
  k_conv3x3<<<1080, 256, 0, stream>>>(padA, w16_h1, hm1_sc, hm1_b,
                                      nullptr, padB, 128, 128, 128, 1, 0, 0);
  k_conv3x3<<<1080, 256, 0, stream>>>(padB, w16_h2, nullptr, hm2bp,
                                      hm, nullptr, 128, 16, 10, 2, 0, 1);

  // ---- NMS + top-200 proposal selection ----
  k_nms<<<cdiv(10 * HWSZ, 256), 256, 0, stream>>>(hm, hmnms, tscr);
  k_topk<<<1, 1024, 0, stream>>>(tscr, cidx, pidx);
  k_build_q<<<cdiv(NPQ * 128, 256), 256, 0, stream>>>(lidar_tm, cls_w, cls_b, cidx, pidx, qt);
  k_build_qpos<<<1, 256, 0, stream>>>(cidx, pidx, hmnms, qpos, qsc);

  auto gemm = [&](const _Float16* W, const float* act, float* o,
                  const float* scale, const float* shift,
                  int M, int N, int K, int lda, int ldc, int relu) {
    int tiles = (M / 16) * ((N + 15) / 16);
    k_gemm<<<cdiv(tiles, 8), 256, 0, stream>>>(W, act, o, scale, shift, M, N, K, lda, ldc, relu);
  };

  // ---- positional embeddings ----
  k_pe1<<<cdiv(NPQ * 128, 256), 256, 0, stream>>>(qpos, ps_w1, ps_b1, ps_sc, ps_b, qpeh, NPQ, 0);
  gemm(g_psw2, qpeh, qpe, nullptr, ps_b2, 128, NPQ, 128, 128, 128, 0);
  k_pe1<<<cdiv(HWSZ * 128, 256), 256, 0, stream>>>(nullptr, pc_w1, pc_b1, pc_sc, pc_b, bevh, HWSZ, 1);
  gemm(g_pcw2, bevh, kpe, nullptr, pc_b2, 128, HWSZ, 128, 128, 128, 0);

  // ---- self-attention ----
  k_vadd<<<cdiv(NPQ * 128, 256), 256, 0, stream>>>(qt, qpe, q0, (size_t)NPQ * 128);
  gemm(g_sawq, q0, sq, nullptr, sa_bq, 128, NPQ, 128, 128, 128, 0);
  gemm(g_sawk, q0, sk, nullptr, sa_bk, 128, NPQ, 128, 128, 128, 0);
  gemm(g_sawv, q0, sv, nullptr, sa_bv, 128, NPQ, 128, 128, 128, 0);
  k_attn<<<cdiv(NPQ, 16), 256, 0, stream>>>(sq, sk, sv, ao, NPQ, NPQ);
  gemm(g_sawo, ao, po, nullptr, sa_bo, 128, NPQ, 128, 128, 128, 0);
  k_add_ln<<<cdiv(NPQ, 8), 256, 0, stream>>>(qt, po, ln1_g, ln1_b, q1, NPQ);

  // ---- cross-attention over 32400 BEV tokens ----
  k_vadd<<<cdiv(NPQ * 128, 256), 256, 0, stream>>>(q1, qpe, qin, (size_t)NPQ * 128);
  k_vadd<<<cdiv(HWSZ * 128, 256), 256, 0, stream>>>(lidar_tm, kpe, kvin, (size_t)HWSZ * 128);
  gemm(g_cawq, qin, cq, nullptr, ca_bq, 128, NPQ, 128, 128, 128, 0);
  gemm(g_cawk, kvin, kproj, nullptr, ca_bk, 128, HWSZ, 128, 128, 128, 0);
  gemm(g_cawv, kvin, vproj, nullptr, ca_bv, 128, HWSZ, 128, 128, 128, 0);
  k_attn<<<cdiv(NPQ, 16), 256, 0, stream>>>(cq, kproj, vproj, ao, NPQ, HWSZ);
  gemm(g_cawo, ao, po, nullptr, ca_bo, 128, NPQ, 128, 128, 128, 0);
  k_add_ln<<<cdiv(NPQ, 8), 256, 0, stream>>>(q1, po, ln2_g, ln2_b, q2, NPQ);

  // ---- FFN ----
  gemm(g_ffn1, q2, f1, nullptr, ffn_b1, 256, NPQ, 128, 128, 256, 1);
  gemm(g_ffn2, f1, f2, nullptr, ffn_b2, 128, NPQ, 256, 256, 128, 0);
  k_add_ln<<<cdiv(NPQ, 8), 256, 0, stream>>>(q2, f2, ln3_g, ln3_b, q3, NPQ);

  // ---- prediction heads ----
  for (int j = 0; j < 6; ++j) {
    gemm(g_hw1[j], q3, hid1, hg_sc[j], h_b[j], 64, NPQ, 128, 128, 64, 1);
    gemm(g_hw2[j], hid1, hout[j], nullptr, hb2p[j], 16, NPQ, 64, 64, 16, 0);
  }

  // ---- decode boxes / scores / labels ----
  k_decode<<<1, 256, 0, stream>>>(hout[0], hout[3], hout[1], hout[4], hout[5],
                                  hout[2], qpos, qsc, cidx, out);
#undef PF
}
